// Attention_49091476193381
// MI455X (gfx1250) — compile-verified, matrix operand order checked by reading, not run on hardware
//
#include <hip/hip_runtime.h>
#include <math.h>
#include <stdint.h>

typedef __attribute__((ext_vector_type(16))) __bf16 v16bf;
typedef __attribute__((ext_vector_type(8)))  float  v8f;

constexpr int Bsz = 2, Hn = 16, Lseq = 2048, Dh = 64, HID = 1024;
constexpr int Mrows = Bsz * Lseq;  // 4096

// K-index of the bf16 pair held in fragment "vgpr" v for lane-half hi
// (CDNA5 ISA 7.12.2, 16-bit A 16x32 layout; B mirrors with lane = N column).
__device__ __forceinline__ int frag_k(int v, int hi) {
  return ((v & 4) << 2) + hi * 8 + 2 * (v & 3);
}

__device__ __forceinline__ v8f wmma_bf16(v16bf a, v16bf b, v8f c) {
  return __builtin_amdgcn_wmma_f32_16x16x32_bf16(false, a, false, b, (short)0, c,
                                                 false, false);
}

// Async copy 16B global -> LDS (CDNA5 GLOBAL_LOAD_ASYNC_TO_LDS_B128, ASYNCcnt).
__device__ __forceinline__ void async_ld_b128(unsigned lds_addr, const void* gptr) {
  unsigned long long ga = (unsigned long long)(uintptr_t)gptr;
  asm volatile("global_load_async_to_lds_b128 %0, %1, off"
               :: "v"(lds_addr), "v"(ga) : "memory");
}
// Async loads complete in order: waiting <=6 guarantees the oldest 6 are done.
__device__ __forceinline__ void wait_asynccnt0() {
  asm volatile("s_wait_asynccnt 0" ::: "memory");
}
__device__ __forceinline__ void wait_asynccnt6() {
  asm volatile("s_wait_asynccnt 6" ::: "memory");
}

// ---------------------------------------------------------------- cast kernel
__global__ void cast_f32_bf16(const float* __restrict__ in, __bf16* __restrict__ out,
                              int n) {
  int i = blockIdx.x * blockDim.x + threadIdx.x;
  if (i < n) out[i] = (__bf16)in[i];
}

// ------------------------------------------------------------------ GEMM
// C[M=4096, N=1024] = A[M,K=1024] * W[N,K]^T   (bf16 in, f32 accumulate)
// Block tile 128x64 (4 waves, 32x64 per wave), K-step 32.
// Double-buffered async-LDS staging: next tile's loads overlap current WMMAs.
// out_mode: 0 = bf16 [M,N]
//           1 = bf16 head-split  [B,H,L,D]
//           2 = bf16 head-split, transposed [B,H,D,L]
//           3 = f32 [M,N] + bias (final output)
__global__ __launch_bounds__(128)
void gemm_bf16_kernel(const __bf16* __restrict__ A, const __bf16* __restrict__ W,
                      float scale, int out_mode, __bf16* __restrict__ outb,
                      float* __restrict__ outf, const float* __restrict__ bias) {
  constexpr int Kdim = HID, N = HID;
  constexpr int LDT = 40;  // padded LDS row stride (bf16 elems)
  constexpr int NSTEP = Kdim / 32;
  __shared__ __align__(16) __bf16 As[2][128 * LDT];
  __shared__ __align__(16) __bf16 Bs[2][64 * LDT];

  const int tid  = threadIdx.x;
  const int lane = tid & 31, wave = tid >> 5;
  const int nl = lane & 15, hi2 = lane >> 4;
  const int m0 = blockIdx.y * 128, n0 = blockIdx.x * 64;

  // Per-thread staging slot: row = tid>>2 (+i*32), kc = (tid&3)*8.
  const int srow = tid >> 2;
  const int skc  = (tid & 3) * 8;

  auto stage = [&](int buf, int k0) {
#pragma unroll
    for (int i = 0; i < 4; ++i) {
      int row = srow + i * 32;
      async_ld_b128((unsigned)(uintptr_t)&As[buf][row * LDT + skc],
                    A + (size_t)(m0 + row) * Kdim + k0 + skc);
    }
#pragma unroll
    for (int i = 0; i < 2; ++i) {
      int row = srow + i * 32;
      async_ld_b128((unsigned)(uintptr_t)&Bs[buf][row * LDT + skc],
                    W + (size_t)(n0 + row) * Kdim + k0 + skc);
    }
  };

  v8f acc[2][4];
#pragma unroll
  for (int a = 0; a < 2; ++a)
#pragma unroll
    for (int j = 0; j < 4; ++j)
#pragma unroll
      for (int e = 0; e < 8; ++e) acc[a][j][e] = 0.f;

  stage(0, 0);

  for (int step = 0; step < NSTEP; ++step) {
    const int buf = step & 1;
    if (step + 1 < NSTEP) {
      stage(buf ^ 1, (step + 1) * 32);  // prefetch next tile (overlaps compute)
      wait_asynccnt6();                 // current tile's 6 oldest loads done
    } else {
      wait_asynccnt0();
    }
    __syncthreads();

    v16bf af[2];
#pragma unroll
    for (int a = 0; a < 2; ++a) {
      const __bf16* rowp = &As[buf][(wave * 32 + a * 16 + nl) * LDT];
#pragma unroll
      for (int v = 0; v < 8; ++v) {
        int k = frag_k(v, hi2);
        af[a][2 * v] = rowp[k];
        af[a][2 * v + 1] = rowp[k + 1];
      }
    }
#pragma unroll
    for (int j = 0; j < 4; ++j) {
      v16bf bfr;
      const __bf16* rowp = &Bs[buf][(j * 16 + nl) * LDT];
#pragma unroll
      for (int v = 0; v < 8; ++v) {
        int k = frag_k(v, hi2);
        bfr[2 * v] = rowp[k];
        bfr[2 * v + 1] = rowp[k + 1];
      }
#pragma unroll
      for (int a = 0; a < 2; ++a) acc[a][j] = wmma_bf16(af[a], bfr, acc[a][j]);
    }
    __syncthreads();  // all waves done reading before this buf is re-staged
  }

  // Epilogue. C layout: vgpr r, lane -> row = r + 8*hi2, col = nl.
#pragma unroll
  for (int a = 0; a < 2; ++a) {
#pragma unroll
    for (int j = 0; j < 4; ++j) {
#pragma unroll
      for (int r = 0; r < 8; ++r) {
        int m = m0 + wave * 32 + a * 16 + r + 8 * hi2;
        int n = n0 + j * 16 + nl;
        float val = acc[a][j][r] * scale;
        if (out_mode == 0) {
          outb[(size_t)m * N + n] = (__bf16)val;
        } else if (out_mode == 1) {
          int b = m >> 11, q = m & (Lseq - 1);
          int hh = n >> 6, d = n & (Dh - 1);
          outb[((((size_t)b * Hn + hh) * Lseq) + q) * Dh + d] = (__bf16)val;
        } else if (out_mode == 2) {
          int b = m >> 11, q = m & (Lseq - 1);
          int hh = n >> 6, d = n & (Dh - 1);
          outb[((((size_t)b * Hn + hh) * Dh) + d) * Lseq + q] = (__bf16)val;
        } else {
          outf[(size_t)m * N + n] = val + bias[n];
        }
      }
    }
  }
}

// ---------------------------------------------------------------- attention
// One wave per 16-row q-tile per (b,h). Flash-style online softmax.
__global__ __launch_bounds__(32)
void attn_kernel(const __bf16* __restrict__ Qh, const __bf16* __restrict__ Kh,
                 const __bf16* __restrict__ Vt, const int* __restrict__ qmask,
                 const int* __restrict__ kmask, __bf16* __restrict__ AO) {
  __shared__ __align__(16) __bf16 p_lds[16 * 32];
  const int lane = threadIdx.x & 31;
  const int nl = lane & 15, hi2 = lane >> 4;

  int flat = blockIdx.x;
  const int QT = Lseq / 16;
  const int qt = flat % QT; flat /= QT;
  const int h = flat % Hn;
  const int b = flat / Hn;
  const int qbase = qt * 16;

  const __bf16* Qp = Qh + (((size_t)(b * Hn + h)) * Lseq + qbase) * Dh;
  const __bf16* Kp = Kh + ((size_t)(b * Hn + h)) * Lseq * Dh;
  const __bf16* Vp = Vt + ((size_t)(b * Hn + h)) * Dh * Lseq;

  // Q fragments (A layout): K-dim = d in [0,32) and [32,64)
  v16bf qa[2];
#pragma unroll
  for (int kk = 0; kk < 2; ++kk) {
    const __bf16* rowp = Qp + nl * Dh + kk * 32;
#pragma unroll
    for (int v = 0; v < 8; ++v) {
      int k = frag_k(v, hi2);
      qa[kk][2 * v] = rowp[k];
      qa[kk][2 * v + 1] = rowp[k + 1];
    }
  }

  int qmr[8];
  float m_r[8], s_r[8];
  v8f o[4];
#pragma unroll
  for (int r = 0; r < 8; ++r) {
    qmr[r] = qmask[b * Lseq + qbase + r + 8 * hi2];
    m_r[r] = -3.0e38f;
    s_r[r] = 0.f;
  }
#pragma unroll
  for (int j = 0; j < 4; ++j)
#pragma unroll
    for (int e = 0; e < 8; ++e) o[j][e] = 0.f;

  const int kend = qbase + 16;  // causal: keys <= qbase+15
  for (int kbase = 0; kbase < kend; kbase += 32) {
    // ----- S = Q K^T for 32 keys (two 16-key N tiles)
    v8f s[2];
#pragma unroll
    for (int t = 0; t < 2; ++t) {
      int key = kbase + t * 16 + nl;
      int keyc = key < Lseq ? key : Lseq - 1;
      const __bf16* rowp = Kp + (size_t)keyc * Dh;
      v16bf kb0, kb1;
#pragma unroll
      for (int v = 0; v < 8; ++v) {
        int k = frag_k(v, hi2);
        kb0[2 * v] = rowp[k];          kb0[2 * v + 1] = rowp[k + 1];
        kb1[2 * v] = rowp[32 + k];     kb1[2 * v + 1] = rowp[32 + k + 1];
      }
      v8f z;
#pragma unroll
      for (int e = 0; e < 8; ++e) z[e] = 0.f;
      s[t] = wmma_bf16(qa[0], kb0, z);
      s[t] = wmma_bf16(qa[1], kb1, s[t]);
    }

    // ----- masking + online softmax
    float p[2][8], rmax[8], rsum[8], alpha[8];
#pragma unroll
    for (int t = 0; t < 2; ++t) {
      int key = kbase + t * 16 + nl;
      int keyc = key < Lseq ? key : Lseq - 1;
      int km = kmask[b * Lseq + keyc];
#pragma unroll
      for (int r = 0; r < 8; ++r) {
        int mrow = qbase + r + 8 * hi2;
        bool ok = (key <= mrow) && (km != 0) && (qmr[r] != 0);
        p[t][r] = s[t][r] + (ok ? 0.f : -1.0e9f);
      }
    }
#pragma unroll
    for (int r = 0; r < 8; ++r) rmax[r] = fmaxf(p[0][r], p[1][r]);
#pragma unroll
    for (int off = 8; off >= 1; off >>= 1)
#pragma unroll
      for (int r = 0; r < 8; ++r)
        rmax[r] = fmaxf(rmax[r], __shfl_xor(rmax[r], off, 32));
#pragma unroll
    for (int r = 0; r < 8; ++r) {
      float mn = fmaxf(m_r[r], rmax[r]);
      alpha[r] = __expf(m_r[r] - mn);
      m_r[r] = mn;
      rsum[r] = 0.f;
    }
#pragma unroll
    for (int t = 0; t < 2; ++t)
#pragma unroll
      for (int r = 0; r < 8; ++r) {
        float e = __expf(p[t][r] - m_r[r]);
        p[t][r] = e;
        rsum[r] += e;
      }
#pragma unroll
    for (int off = 8; off >= 1; off >>= 1)
#pragma unroll
      for (int r = 0; r < 8; ++r) rsum[r] += __shfl_xor(rsum[r], off, 32);
#pragma unroll
    for (int r = 0; r < 8; ++r) s_r[r] = s_r[r] * alpha[r] + rsum[r];
#pragma unroll
    for (int j = 0; j < 4; ++j)
#pragma unroll
      for (int r = 0; r < 8; ++r) o[j][r] *= alpha[r];

    // ----- C-layout P -> LDS -> A-layout fragment
#pragma unroll
    for (int t = 0; t < 2; ++t)
#pragma unroll
      for (int r = 0; r < 8; ++r)
        p_lds[(r + 8 * hi2) * 32 + t * 16 + nl] = (__bf16)p[t][r];
    __syncthreads();
    v16bf pa;
    {
      const __bf16* rowp = &p_lds[nl * 32];
#pragma unroll
      for (int v = 0; v < 8; ++v) {
        int k = frag_k(v, hi2);
        pa[2 * v] = rowp[k];
        pa[2 * v + 1] = rowp[k + 1];
      }
    }
    __syncthreads();

    // ----- O += P * V (V stored [B,H,D,L] so B-columns are contiguous rows)
#pragma unroll
    for (int j = 0; j < 4; ++j) {
      v16bf vb;
      const __bf16* rowp = Vp + (size_t)(j * 16 + nl) * Lseq;
#pragma unroll
      for (int v = 0; v < 8; ++v) {
        int k = frag_k(v, hi2);
        int k0c = kbase + k;     if (k0c >= Lseq) k0c = Lseq - 1;
        int k1c = kbase + k + 1; if (k1c >= Lseq) k1c = Lseq - 1;
        vb[2 * v] = rowp[k0c];
        vb[2 * v + 1] = rowp[k1c];
      }
      o[j] = wmma_bf16(pa, vb, o[j]);
    }
  }

  // Epilogue: normalize and write [B, L, H*D] bf16 for the output projection.
#pragma unroll
  for (int j = 0; j < 4; ++j)
#pragma unroll
    for (int r = 0; r < 8; ++r) {
      float val = o[j][r] / s_r[r];
      int mrow = qbase + r + 8 * hi2;
      int col = h * Dh + j * 16 + nl;
      AO[((size_t)(b * Lseq + mrow)) * HID + col] = (__bf16)val;
    }
}

// ---------------------------------------------------------------- launcher
extern "C" void kernel_launch(void* const* d_in, const int* in_sizes, int n_in,
                              void* d_out, int out_size, void* d_ws, size_t ws_size,
                              hipStream_t stream) {
  const float* x  = (const float*)d_in[0];
  const int*   qm = (const int*)d_in[1];
  const int*   km = (const int*)d_in[2];
  const float* Wq = (const float*)d_in[3];
  const float* Wk = (const float*)d_in[4];
  const float* Wv = (const float*)d_in[5];
  const float* Wo = (const float*)d_in[6];
  const float* bo = (const float*)d_in[7];
  float* out = (float*)d_out;

  char* ws = (char*)d_ws;
  size_t off = 0;
  auto alloc = [&](size_t bytes) -> char* {
    char* p = ws + off;
    off += (bytes + 255) & ~(size_t)255;
    return p;
  };
  __bf16* xb  = (__bf16*)alloc((size_t)Mrows * HID * 2);
  __bf16* wqb = (__bf16*)alloc((size_t)HID * HID * 2);
  __bf16* wkb = (__bf16*)alloc((size_t)HID * HID * 2);
  __bf16* wvb = (__bf16*)alloc((size_t)HID * HID * 2);
  __bf16* wob = (__bf16*)alloc((size_t)HID * HID * 2);
  __bf16* Qh  = (__bf16*)alloc((size_t)Bsz * Hn * Lseq * Dh * 2);
  __bf16* Kh  = (__bf16*)alloc((size_t)Bsz * Hn * Lseq * Dh * 2);
  __bf16* Vt  = (__bf16*)alloc((size_t)Bsz * Hn * Dh * Lseq * 2);
  __bf16* AO  = (__bf16*)alloc((size_t)Mrows * HID * 2);

  int nX = Mrows * HID;
  cast_f32_bf16<<<(nX + 255) / 256, 256, 0, stream>>>(x, xb, nX);
  int nW = HID * HID;
  cast_f32_bf16<<<(nW + 255) / 256, 256, 0, stream>>>(Wq, wqb, nW);
  cast_f32_bf16<<<(nW + 255) / 256, 256, 0, stream>>>(Wk, wkb, nW);
  cast_f32_bf16<<<(nW + 255) / 256, 256, 0, stream>>>(Wv, wvb, nW);
  cast_f32_bf16<<<(nW + 255) / 256, 256, 0, stream>>>(Wo, wob, nW);

  dim3 grid(HID / 64, Mrows / 128);
  // Q scaled by 1/sqrt(D) = 0.125
  gemm_bf16_kernel<<<grid, 128, 0, stream>>>(xb, wqb, 0.125f, 1, Qh, nullptr, nullptr);
  gemm_bf16_kernel<<<grid, 128, 0, stream>>>(xb, wkb, 1.0f,  1, Kh, nullptr, nullptr);
  gemm_bf16_kernel<<<grid, 128, 0, stream>>>(xb, wvb, 1.0f,  2, Vt, nullptr, nullptr);

  attn_kernel<<<Bsz * Hn * (Lseq / 16), 32, 0, stream>>>(Qh, Kh, Vt, qm, km, AO);

  gemm_bf16_kernel<<<grid, 128, 0, stream>>>(AO, wob, 1.0f, 3, nullptr, out, bo);
}
